// SSD_ICGA_65214783423070
// MI455X (gfx1250) — matrix operand order repaired and checked by reading, not compiled, not run on hardware
//
#include <hip/hip_runtime.h>
#include <hip/hip_bf16.h>

// ---------------------------------------------------------------------------
// Types for CDNA5 WMMA
// ---------------------------------------------------------------------------
typedef __attribute__((ext_vector_type(16))) __bf16 v16bf;
typedef __attribute__((ext_vector_type(8)))  __bf16 v8bf;
typedef __attribute__((ext_vector_type(8)))  float  v8f;

#define LDSS 40   // LDS row stride in bf16 elements (80B: 16B aligned, conflict-staggered)

// Async global->LDS path (GLOBAL_LOAD_ASYNC_TO_LDS_B128, tracked by ASYNCcnt).
// Builtin signature (from compiler diagnostic): (int4 AS1*, int4 AS3*, imm, imm)
#if __has_builtin(__builtin_amdgcn_global_load_async_to_lds_b128) && \
    __has_builtin(__builtin_amdgcn_s_wait_asynccnt)
#define HAVE_ASYNC_LDS 1
typedef int v4i128 __attribute__((vector_size(4 * sizeof(int))));
#define AS1P(p) ((__attribute__((address_space(1))) v4i128*)(p))
#define AS3P(p) ((__attribute__((address_space(3))) v4i128*)(p))
#else
#define HAVE_ASYNC_LDS 0
#endif

__device__ __forceinline__ __bf16 f2bf(float x) {
    unsigned u = __builtin_bit_cast(unsigned, x);
    unsigned r = u + 0x7FFFu + ((u >> 16) & 1u);      // round-to-nearest-even
    unsigned short h = (unsigned short)(r >> 16);
    return __builtin_bit_cast(__bf16, h);
}

// ---------------------------------------------------------------------------
// Tiled bf16 WMMA GEMM: C[M,N] = A[M,K] * B[K,N], all row-major, C in f32.
// Block = 128 threads = 4 wave32s. Block tile 64x64, K-step 32.
// Wave w computes rows [64*by + 16w, +16) x cols [64*bx, +64) as 4 WMMA tiles.
// Requires: M % 64 == 0, N % 16 == 0 (N >= 64 per block tile), K % 32 == 0.
// ---------------------------------------------------------------------------
__global__ __launch_bounds__(128)
void k_gemm_bf16(const __bf16* __restrict__ A, const __bf16* __restrict__ B,
                 float* __restrict__ C, int M, int N, int K) {
    __shared__ __attribute__((aligned(16))) __bf16 ldsA[64 * LDSS];   // [m][k]
    __shared__ __attribute__((aligned(16))) __bf16 ldsB[64 * LDSS];   // transposed: [n][k]

    const int tid  = threadIdx.x;
    const int wave = tid >> 5;
    const int lane = tid & 31;
    const int half = (lane < 16) ? 0 : 1;   // which 16-lane half of the wave
    const int l16  = lane & 15;
    const int m0   = blockIdx.y * 64;
    const int n0   = blockIdx.x * 64;

    // A staging map: thread -> (row, 16-half segment)
    const int ar   = tid >> 1;
    const int aoff = (tid & 1) * 16;
    // B staging map: 4x4 register micro-transpose; thread -> (k-quad, n-quad)
    const int kq = (tid & 7) * 4;
    const int nq = (tid >> 3) * 4;

    v8f acc[4] = {v8f{}, v8f{}, v8f{}, v8f{}};

    for (int kk = 0; kk < K; kk += 32) {
        // --- stage A panel (64 x 32) into LDS: two 16B transfers per thread ---
        {
            const __bf16* src = A + (size_t)(m0 + ar) * K + kk + aoff;
            __bf16* dst = &ldsA[ar * LDSS + aoff];
#if HAVE_ASYNC_LDS
            __builtin_amdgcn_global_load_async_to_lds_b128(AS1P(src),     AS3P(dst),     0, 0);
            __builtin_amdgcn_global_load_async_to_lds_b128(AS1P(src + 8), AS3P(dst + 8), 0, 0);
#else
            *(uint4*)dst       = *(const uint4*)(src);
            *(uint4*)(dst + 8) = *(const uint4*)(src + 8);
#endif
        }
        // --- stage B panel (32 x 64) transposed into LDS [n][k]:
        //     4x4 micro-transpose in registers, 4x b64 loads + 4x ds_store_b64 ---
        {
            __bf16 t4[4][4] __attribute__((aligned(8)));
            #pragma unroll
            for (int i = 0; i < 4; ++i)
                *(uint2*)&t4[i][0] = *(const uint2*)(B + (size_t)(kk + kq + i) * N + n0 + nq);
            #pragma unroll
            for (int j = 0; j < 4; ++j) {
                __bf16 col[4] __attribute__((aligned(8))) =
                    { t4[0][j], t4[1][j], t4[2][j], t4[3][j] };
                *(uint2*)&ldsB[(nq + j) * LDSS + kq] = *(const uint2*)col;
            }
        }
#if HAVE_ASYNC_LDS
        __builtin_amdgcn_s_wait_asynccnt(0);
#endif
        __syncthreads();

        if (kk + 32 < K) {   // pull next panels toward the caches (global_prefetch)
            __builtin_prefetch(A + (size_t)(m0 + ar) * K + kk + 32, 0, 1);
            __builtin_prefetch(B + (size_t)(kk + 32 + kq) * N + n0, 0, 1);
        }

        // --- A fragment (16x32) per ISA layout:
        // lanes 0-15: halves 0-7 <- k 0..7,  halves 8-15 <- k 16..23
        // lanes16-31: halves 0-7 <- k 8..15, halves 8-15 <- k 24..31
        const __bf16* ap = &ldsA[(wave * 16 + l16) * LDSS + half * 8];
        v8bf alo = *(const v8bf*)(ap);
        v8bf ahi = *(const v8bf*)(ap + 16);
        v16bf afrag = __builtin_shufflevector(alo, ahi,
            0, 1, 2, 3, 4, 5, 6, 7, 8, 9, 10, 11, 12, 13, 14, 15);

        // --- gather all 4 B fragments first so the WMMAs issue back-to-back ---
        v16bf bfrag[4];
        #pragma unroll
        for (int t = 0; t < 4; ++t) {
            // B fragment (32x16): lane holds column n = 16t + l16;
            // lanes 0-15 take k 0..7,16..23, lanes 16-31 take k 8..15,24..31 (same order as A)
            const __bf16* bp = &ldsB[(t * 16 + l16) * LDSS + half * 8];
            v8bf blo = *(const v8bf*)(bp);
            v8bf bhi = *(const v8bf*)(bp + 16);
            bfrag[t] = __builtin_shufflevector(blo, bhi,
                0, 1, 2, 3, 4, 5, 6, 7, 8, 9, 10, 11, 12, 13, 14, 15);
        }
        #pragma unroll
        for (int t = 0; t < 4; ++t)
            acc[t] = __builtin_amdgcn_wmma_f32_16x16x32_bf16(
                false, afrag, false, bfrag[t], (short)0, acc[t], false, false);

        __syncthreads();
    }

    // --- epilogue: C/D layout: VGPR r -> M = r + 8*half, lane -> N = l16 ---
    #pragma unroll
    for (int t = 0; t < 4; ++t) {
        const int n = n0 + t * 16 + l16;
        #pragma unroll
        for (int r = 0; r < 8; ++r) {
            const int m = m0 + wave * 16 + half * 8 + r;
            if (m < M && n < N) C[(size_t)m * N + n] = acc[t][r];
        }
    }
}

// ---------------------------------------------------------------------------
// Utility kernels
// ---------------------------------------------------------------------------
__global__ void k_zero(float* p, size_t n) {
    for (size_t i = (size_t)blockIdx.x * blockDim.x + threadIdx.x; i < n;
         i += (size_t)gridDim.x * blockDim.x) p[i] = 0.f;
}
__global__ void k_copy(float* __restrict__ d, const float* __restrict__ s, size_t n) {
    for (size_t i = (size_t)blockIdx.x * blockDim.x + threadIdx.x; i < n;
         i += (size_t)gridDim.x * blockDim.x) d[i] = s[i];
}
__global__ void k_add(float* __restrict__ d, const float* __restrict__ s, size_t n) {
    for (size_t i = (size_t)blockIdx.x * blockDim.x + threadIdx.x; i < n;
         i += (size_t)gridDim.x * blockDim.x) d[i] += s[i];
}
__global__ void k_scale(float* __restrict__ d, const float* __restrict__ s, float a, size_t n) {
    for (size_t i = (size_t)blockIdx.x * blockDim.x + threadIdx.x; i < n;
         i += (size_t)gridDim.x * blockDim.x) d[i] = s[i] * a;
}
__global__ void k_f32_to_bf16(__bf16* __restrict__ d, const float* __restrict__ s, size_t n) {
    for (size_t i = (size_t)blockIdx.x * blockDim.x + threadIdx.x; i < n;
         i += (size_t)gridDim.x * blockDim.x) d[i] = f2bf(s[i]);
}

// LightGCN SpMM scatter: y[row] += val * x[col], 64 lanes per edge (D=64)
__global__ void k_spmm(const int* __restrict__ rows, const int* __restrict__ cols,
                       const float* __restrict__ vals, const float* __restrict__ x,
                       float* __restrict__ y, int ne) {
    const size_t idx = (size_t)blockIdx.x * blockDim.x + threadIdx.x;
    const int e = (int)(idx >> 6);
    const int t = (int)(idx & 63);
    if (e < ne) {
        const float v = vals[e] * x[(size_t)cols[e] * 64 + t];
        atomicAdd(&y[(size_t)rows[e] * 64 + t], v);
    }
}

// Edge MLP: hid = relu(P[col] + Q[row] + b1); p = hid.w2 + b2; A[row,col] += sigmoid(p)
__global__ __launch_bounds__(256)
void k_edge(const int* __restrict__ rows, const int* __restrict__ cols,
            const float* __restrict__ P, const float* __restrict__ Q,
            const float* __restrict__ b1, const float* __restrict__ w2,
            const float* __restrict__ b2, float* __restrict__ Adense, int ne) {
    const int e    = blockIdx.x * 8 + (threadIdx.x >> 5);
    const int lane = threadIdx.x & 31;
    if (e >= ne) return;
    const int r = rows[e], c = cols[e];
    float h0 = P[(size_t)c * 64 + lane]      + Q[(size_t)r * 64 + lane]      + b1[lane];
    float h1 = P[(size_t)c * 64 + lane + 32] + Q[(size_t)r * 64 + lane + 32] + b1[lane + 32];
    h0 = fmaxf(h0, 0.f);
    h1 = fmaxf(h1, 0.f);
    float part = h0 * w2[lane] + h1 * w2[lane + 32];
    #pragma unroll
    for (int off = 16; off > 0; off >>= 1) part += __shfl_xor(part, off, 32);
    if (lane == 0) {
        const float p = part + b2[0];
        const float w = 1.f / (1.f + expf(-p));
        atomicAdd(&Adense[(size_t)r * 4096 + c], w);
    }
}

// Masked row softmax (mask = g > 0; rows with no positives -> 0), output bf16
__global__ __launch_bounds__(128)
void k_softmax_bf16(const float* __restrict__ G, __bf16* __restrict__ S, int L) {
    const int row = blockIdx.x;
    const float* g = G + (size_t)row * L;
    __bf16* s = S + (size_t)row * L;
    __shared__ float red[128];
    const int tid = threadIdx.x;

    float mx = -1e30f;
    for (int j = tid; j < L; j += 128) {
        const float v = g[j];
        if (v > 0.f) mx = fmaxf(mx, v);
    }
    red[tid] = mx; __syncthreads();
    for (int o = 64; o > 0; o >>= 1) {
        if (tid < o) red[tid] = fmaxf(red[tid], red[tid + o]);
        __syncthreads();
    }
    mx = red[0]; __syncthreads();

    if (mx <= 0.f) {  // no positive entry in this row
        for (int j = tid; j < L; j += 128) s[j] = f2bf(0.f);
        return;
    }
    float sum = 0.f;
    for (int j = tid; j < L; j += 128) {
        const float v = g[j];
        if (v > 0.f) sum += expf(v - mx);
    }
    red[tid] = sum; __syncthreads();
    for (int o = 64; o > 0; o >>= 1) {
        if (tid < o) red[tid] += red[tid + o];
        __syncthreads();
    }
    const float inv = 1.f / red[0];
    for (int j = tid; j < L; j += 128) {
        const float v = g[j];
        s[j] = f2bf((v > 0.f) ? expf(v - mx) * inv : 0.f);
    }
}

// out = user_e + (user_e + T1 + T2)/3
__global__ void k_final(const float* __restrict__ ue, const float* __restrict__ t1,
                        const float* __restrict__ t2, float* __restrict__ out, size_t n) {
    for (size_t i = (size_t)blockIdx.x * blockDim.x + threadIdx.x; i < n;
         i += (size_t)gridDim.x * blockDim.x)
        out[i] = ue[i] + (ue[i] + t1[i] + t2[i]) * (1.f / 3.f);
}

// ---------------------------------------------------------------------------
// Launcher
// ---------------------------------------------------------------------------
extern "C" void kernel_launch(void* const* d_in, const int* in_sizes, int n_in,
                              void* d_out, int out_size, void* d_ws, size_t ws_size,
                              hipStream_t stream) {
    (void)in_sizes; (void)n_in; (void)out_size; (void)ws_size;
    const int  U = 4096, D = 64, NTOT = 20480;   // N = U + I
    const int  NE_I = 524288, NE_S = 65536;

    const float* user_emb  = (const float*)d_in[0];
    const float* item_emb  = (const float*)d_in[1];
    const float* ivals     = (const float*)d_in[2];
    const float* Wa        = (const float*)d_in[3];
    const float* Wb        = (const float*)d_in[4];
    const float* w1        = (const float*)d_in[5];
    const float* b1        = (const float*)d_in[6];
    const float* w2        = (const float*)d_in[7];
    const float* b2        = (const float*)d_in[8];
    const int*   irows     = (const int*)d_in[9];
    const int*   icols     = (const int*)d_in[10];
    const int*   srows     = (const int*)d_in[11];
    const int*   scols     = (const int*)d_in[12];
    float* out = (float*)d_out;

    // ---- workspace carve-up ----
    char* w = (char*)d_ws;
    const size_t MB = 1024ull * 1024ull;
    float*  Adense = (float*)(w + 0);            // 64 MB  (later reused as S2_bf region)
    float*  G2     = (float*)(w + 64 * MB);      // 64 MB
    __bf16* Abf    = (__bf16*)(w + 128 * MB);    // 32 MB  (later reused as S1_bf)
    char* misc = w + 160 * MB;
    float*  h0   = (float*)misc;                 misc += (size_t)NTOT * D * 4;
    float*  h1   = (float*)misc;                 misc += (size_t)NTOT * D * 4;
    float*  accN = (float*)misc;                 misc += (size_t)NTOT * D * 4;
    float*  ueF  = (float*)misc;                 misc += (size_t)U * D * 4;
    __bf16* ueBf = (__bf16*)misc;                misc += (size_t)U * D * 2;
    __bf16* ue0B = (__bf16*)misc;                misc += (size_t)U * D * 2;
    __bf16* WaB  = (__bf16*)misc;                misc += (size_t)D * D * 2;
    __bf16* WbB  = (__bf16*)misc;                misc += (size_t)D * D * 2;
    __bf16* w1B  = (__bf16*)misc;                misc += (size_t)2 * D * D * 2;
    float*  Ma   = (float*)misc;                 misc += (size_t)D * D * 4;
    float*  Mb   = (float*)misc;                 misc += (size_t)D * D * 4;
    __bf16* MaB  = (__bf16*)misc;                misc += (size_t)D * D * 2;
    __bf16* MbB  = (__bf16*)misc;                misc += (size_t)D * D * 2;
    float*  P    = (float*)misc;                 misc += (size_t)U * D * 4;
    float*  Q    = (float*)misc;                 misc += (size_t)U * D * 4;
    float*  T1   = (float*)misc;                 misc += (size_t)U * D * 4;
    float*  T2   = (float*)misc;                 misc += (size_t)U * D * 4;
    __bf16* S1B  = Abf;                          // reuse after G2 GEMM
    __bf16* S2B  = (__bf16*)Adense;              // reuse after S1 softmax

    const size_t nUD  = (size_t)U * D;
    const size_t nND  = (size_t)NTOT * D;
    const size_t nAA  = 4096ull * 4096ull;

    // ===== 1. LightGCN on interaction graph =====
    k_copy<<<2048, 256, 0, stream>>>(h0, user_emb, nUD);
    k_copy<<<4096, 256, 0, stream>>>(h0 + nUD, item_emb, (size_t)(NTOT - U) * D);
    k_copy<<<4096, 256, 0, stream>>>(accN, h0, nND);
    float* hc = h0; float* hn = h1;
    for (int layer = 0; layer < 3; ++layer) {
        k_zero<<<4096, 256, 0, stream>>>(hn, nND);
        k_spmm<<<(NE_I * 64) / 256, 256, 0, stream>>>(irows, icols, ivals, hc, hn, NE_I);
        k_add<<<4096, 256, 0, stream>>>(accN, hn, nND);
        float* tmp = hc; hc = hn; hn = tmp;
    }
    k_scale<<<2048, 256, 0, stream>>>(ueF, accN, 0.25f, nUD);   // user_e

    // ===== 2. Edge MLP (folded): Ma=Wa@w1_top, Mb=Wb@w1_bot; P=ue0@Ma, Q=ue0@Mb =====
    k_f32_to_bf16<<<64, 256, 0, stream>>>(WaB, Wa, (size_t)D * D);
    k_f32_to_bf16<<<64, 256, 0, stream>>>(WbB, Wb, (size_t)D * D);
    k_f32_to_bf16<<<64, 256, 0, stream>>>(w1B, w1, (size_t)2 * D * D);
    k_f32_to_bf16<<<1024, 256, 0, stream>>>(ue0B, user_emb, nUD);
    k_gemm_bf16<<<dim3(1, 1), 128, 0, stream>>>(WaB, w1B, Ma, D, D, D);
    k_gemm_bf16<<<dim3(1, 1), 128, 0, stream>>>(WbB, w1B + (size_t)D * D, Mb, D, D, D);
    k_f32_to_bf16<<<64, 256, 0, stream>>>(MaB, Ma, (size_t)D * D);
    k_f32_to_bf16<<<64, 256, 0, stream>>>(MbB, Mb, (size_t)D * D);
    k_gemm_bf16<<<dim3(1, U / 64), 128, 0, stream>>>(ue0B, MaB, P, U, D, D);
    k_gemm_bf16<<<dim3(1, U / 64), 128, 0, stream>>>(ue0B, MbB, Q, U, D, D);

    // ===== 3. Dense social adjacency A via per-edge sigmoid scatter =====
    k_zero<<<8192, 256, 0, stream>>>(Adense, nAA);
    k_edge<<<NE_S / 8, 256, 0, stream>>>(srows, scols, P, Q, b1, w2, b2, Adense, NE_S);

    // ===== 4. G2 = A @ A  (bf16 WMMA; mask2 == (G2 > 0) since all A entries > 0) =====
    k_f32_to_bf16<<<8192, 256, 0, stream>>>(Abf, Adense, nAA);
    k_gemm_bf16<<<dim3(64, 64), 128, 0, stream>>>(Abf, Abf, G2, 4096, 4096, 4096);

    // ===== 5. Masked row softmax of A and G2 (mask = g > 0), bf16 outputs =====
    k_softmax_bf16<<<4096, 128, 0, stream>>>(Adense, S1B, 4096);   // into old Abf region
    k_softmax_bf16<<<4096, 128, 0, stream>>>(G2, S2B, 4096);       // into old Adense region

    // ===== 6. T1 = S1 @ user_e, T2 = S2 @ user_e (bf16 WMMA, K = 4096) =====
    k_f32_to_bf16<<<1024, 256, 0, stream>>>(ueBf, ueF, nUD);
    k_gemm_bf16<<<dim3(1, U / 64), 128, 0, stream>>>(S1B, ueBf, T1, U, D, 4096);
    k_gemm_bf16<<<dim3(1, U / 64), 128, 0, stream>>>(S2B, ueBf, T2, U, D, 4096);

    // ===== 7. out = user_e + mean(user_e, T1, T2) =====
    k_final<<<2048, 256, 0, stream>>>(ueF, T1, T2, out, nUD);
}
